// LuongGlobalAttentionDecoderSeq_91010357002640
// MI455X (gfx1250) — compile-verified, matrix operand order checked by reading, not run on hardware
//
#include <hip/hip_runtime.h>
#include <hip/hip_bf16.h>

// ---------------------------------------------------------------------------
// LuongGlobalAttentionDecoderSeq for MI455X (gfx1250, wave32, WMMA bf16)
//   V=32000, H=1024, E=512, ST=2048, SS=4096
// ---------------------------------------------------------------------------
constexpr int Vv  = 32000;
constexpr int Hd  = 1024;
constexpr int Ed  = 512;
constexpr int STt = 2048;
constexpr int SSs = 4096;

constexpr int GRU_BLOCKS  = 32;   // persistent cooperative grid
constexpr int GRU_THREADS = 256;  // 8 waves
constexpr int GRU_JB      = Hd / GRU_BLOCKS;   // 32 hidden indices per block
constexpr int GRU_ROWS    = 3 * GRU_JB;        // 96 w_hh rows per block

typedef __attribute__((ext_vector_type(16))) __bf16 v16bf;
typedef __attribute__((ext_vector_type(8)))  __bf16 v8bf;
typedef __attribute__((ext_vector_type(8)))  float  v8f;

// float -> bf16 round-to-nearest-even
__device__ inline __bf16 f2bf(float f) {
    union { unsigned short s; __bf16 b; } u;
    unsigned int x = __float_as_uint(f);
    u.s = (unsigned short)((x + 0x7FFFu + ((x >> 16) & 1u)) >> 16);
    return u.b;
}

// Load one lane's 16-element slice of a 16x32 bf16 WMMA fragment.
// ISA 16-bit layout: lanes 0-15 hold K=base+0..7 / base+16..23,
// lanes 16-31 hold K=base+8..15 / base+24..31 (caller adds kSel).
__device__ inline v16bf load_frag16(const __bf16* __restrict__ p) {
    v8bf lo = *(const v8bf*)(p);
    v8bf hi = *(const v8bf*)(p + 16);
    v16bf f;
#pragma unroll
    for (int i = 0; i < 8; ++i) { f[i] = lo[i]; f[i + 8] = hi[i]; }
    return f;
}

// ---------------------------------------------------------------------------
// elementwise f32 -> bf16 convert
// ---------------------------------------------------------------------------
__global__ void cvt_f32_bf16_kernel(const float* __restrict__ src,
                                    __bf16* __restrict__ dst, long long n) {
    long long stride = (long long)gridDim.x * blockDim.x;
    for (long long i = blockIdx.x * (long long)blockDim.x + threadIdx.x; i < n; i += stride)
        dst[i] = f2bf(src[i]);
}

// emb = relu(emb_table[x]) -> bf16 (ST x E)
__global__ void embed_relu_kernel(const int* __restrict__ x,
                                  const float* __restrict__ emb_table,
                                  __bf16* __restrict__ embB) {
    long long n = (long long)STt * Ed;
    long long stride = (long long)gridDim.x * blockDim.x;
    for (long long i = blockIdx.x * (long long)blockDim.x + threadIdx.x; i < n; i += stride) {
        int t = (int)(i >> 9);
        int e = (int)(i & (Ed - 1));
        float v = emb_table[(size_t)x[t] * Ed + e];
        embB[i] = f2bf(v > 0.0f ? v : 0.0f);
    }
}

// ---------------------------------------------------------------------------
// Register-tiled WMMA GEMM:
//   C(f32, MxN) = A(bf16, MxK rm) * BT(bf16, NxK rm)^T + bias[col]
// Block tile: 64(M) x 256(N); wave tile 64x64 = 4x4 WMMA tiles.
// 16 v_wmma per K-step; 4x A/B fragment reuse.
// ---------------------------------------------------------------------------
__global__ __launch_bounds__(128)
void wmma_gemm_bf16_kernel(const __bf16* __restrict__ A,
                           const __bf16* __restrict__ BT,
                           const float* __restrict__ bias,
                           float* __restrict__ C,
                           int M, int N, int K,
                           int lda, int ldb, int ldc) {
    const int lane  = threadIdx.x & 31;
    const int wave  = threadIdx.x >> 5;
    const int mBase = blockIdx.y * 64;
    const int nBase = blockIdx.x * 256 + wave * 64;
    if (mBase >= M || nBase >= N) return;

    const int half16 = lane >> 4;
    const int kSel   = half16 * 8;
    const int rsel   = lane & 15;

    const __bf16* aPtr[4];
    const __bf16* bPtr[4];
#pragma unroll
    for (int i = 0; i < 4; ++i) {
        aPtr[i] = A  + (size_t)(mBase + 16 * i + rsel) * lda;
        bPtr[i] = BT + (size_t)(nBase + 16 * i + rsel) * ldb;
    }

    v8f acc[4][4];
#pragma unroll
    for (int i = 0; i < 4; ++i)
#pragma unroll
        for (int j = 0; j < 4; ++j) acc[i][j] = {};

    for (int k0 = 0; k0 < K; k0 += 32) {
        if (k0 + 64 < K) {  // gfx1250 global_prefetch_b8
            __builtin_prefetch(aPtr[0] + k0 + 64, 0, 3);
            __builtin_prefetch(bPtr[0] + k0 + 64, 0, 3);
        }
        v16bf a[4], b[4];
#pragma unroll
        for (int i = 0; i < 4; ++i) a[i] = load_frag16(aPtr[i] + k0 + kSel);
#pragma unroll
        for (int j = 0; j < 4; ++j) b[j] = load_frag16(bPtr[j] + k0 + kSel);
#pragma unroll
        for (int i = 0; i < 4; ++i)
#pragma unroll
            for (int j = 0; j < 4; ++j)
                acc[i][j] = __builtin_amdgcn_wmma_f32_16x16x32_bf16(
                    false, a[i], false, b[j], (short)0, acc[i][j], false, false);
    }

    // C/D layout: VGPR r -> row 16i + 8*half16 + r, col 16j + lane%16
#pragma unroll
    for (int j = 0; j < 4; ++j) {
        const int col = nBase + 16 * j + rsel;
        const float bv = bias ? bias[col] : 0.0f;
#pragma unroll
        for (int i = 0; i < 4; ++i) {
            const int rBase = mBase + 16 * i + 8 * half16;
#pragma unroll
            for (int r = 0; r < 8; ++r)
                C[(size_t)(rBase + r) * ldc + col] = acc[i][j][r] + bv;
        }
    }
}

// ---------------------------------------------------------------------------
// Persistent cooperative GRU scan across 32 WGPs.
// Each block owns 32 hidden indices -> 96 rows of w_hh, LDS-resident as bf16
// (192 KB out of the WGP's 320 KB), loaded ONCE for the whole 2048-step scan.
// Per step: LDS dot products -> write 32-elem h slice to ping-pong global
// buffer -> grid barrier (L2 atomics + s_sleep spin) -> reload full h to LDS.
// ---------------------------------------------------------------------------
__global__ void gru_sync_init_kernel(unsigned* __restrict__ sync) {
    if (threadIdx.x < 2) sync[threadIdx.x] = 0u;   // [0]=count, [1]=gen
}

__global__ __launch_bounds__(GRU_THREADS)
void gru_persistent_kernel(const float* __restrict__ gi,
                           const float* __restrict__ w_hh,
                           const float* __restrict__ b_hh,
                           float* __restrict__ gru_out,
                           float* __restrict__ hbufA,
                           float* __restrict__ hbufB,
                           unsigned* __restrict__ sync,
                           float* __restrict__ hidden_out) {
    extern __shared__ char smem[];
    __bf16* wlds  = (__bf16*)smem;                                   // 96*1024 bf16
    float*  h_lds = (float*)(smem + (size_t)GRU_ROWS * Hd * 2);      // 1024 f32
    float*  gh    = h_lds + Hd;                                      // 96 f32

    const int tid  = threadIdx.x;
    const int blk  = blockIdx.x;
    const int lane = tid & 31;
    const int wave = tid >> 5;          // 8 waves -> 12 rows each

    // one-time weight load: w_hh rows {g*H + blk*32 + jj} -> LDS bf16
    for (int idx = tid; idx < GRU_ROWS * Hd; idx += GRU_THREADS) {
        int rr = idx >> 10, k = idx & (Hd - 1);
        int g = rr >> 5, jj = rr & 31;
        wlds[idx] = f2bf(w_hh[(size_t)(g * Hd + blk * GRU_JB + jj) * Hd + k]);
    }
    for (int i = tid; i < Hd; i += GRU_THREADS) h_lds[i] = 0.0f;

    float br = 0.0f, bz = 0.0f, bn = 0.0f;
    const int jglob = blk * GRU_JB + tid;
    if (tid < GRU_JB) {
        br = b_hh[jglob];
        bz = b_hh[Hd + jglob];
        bn = b_hh[2 * Hd + jglob];
    }
    __syncthreads();

    float* bufs[2] = { hbufA, hbufB };

    for (int t = 0; t < STt; ++t) {
        // cache this lane's 32 h values (k = lane*2 + 64m, +1)
        float hr[32];
#pragma unroll
        for (int m = 0; m < 16; ++m) {
            float2 hv = *(const float2*)(h_lds + lane * 2 + 64 * m);
            hr[2 * m] = hv.x; hr[2 * m + 1] = hv.y;
        }
        // 12 rows per wave, wave-wide reduction
        for (int i = 0; i < 12; ++i) {
            const int rr = wave * 12 + i;
            const __bf16* wrow = wlds + ((size_t)rr << 10);
            float acc = 0.0f;
#pragma unroll
            for (int m = 0; m < 16; ++m) {
                unsigned p = *(const unsigned*)(wrow + lane * 2 + 64 * m);
                float w0 = __uint_as_float(p << 16);
                float w1 = __uint_as_float(p & 0xFFFF0000u);
                acc += w0 * hr[2 * m] + w1 * hr[2 * m + 1];
            }
            for (int off = 16; off > 0; off >>= 1)
                acc += __shfl_down(acc, off, 32);
            if (lane == 0) gh[rr] = acc;
        }
        __syncthreads();

        // gate update for this block's 32 hidden indices
        float* wbuf = bufs[(t + 1) & 1];
        if (tid < GRU_JB) {
            const float* git = gi + (size_t)t * (3 * Hd);
            float r = 1.0f / (1.0f + __expf(-(git[jglob] + gh[tid] + br)));
            float z = 1.0f / (1.0f + __expf(-(git[Hd + jglob] + gh[GRU_JB + tid] + bz)));
            float n = tanhf(git[2 * Hd + jglob] + r * (gh[2 * GRU_JB + tid] + bn));
            float hnew = (1.0f - z) * n + z * h_lds[jglob];
            wbuf[jglob] = hnew;
            gru_out[(size_t)t * Hd + jglob] = hnew;
        }
        __threadfence();

        // grid barrier (sense via generation counter)
        __syncthreads();
        if (tid == 0) {
            unsigned g = atomicAdd(&sync[1], 0u);
            unsigned a = atomicAdd(&sync[0], 1u);
            if (a == GRU_BLOCKS - 1) {
                atomicExch(&sync[0], 0u);
                atomicAdd(&sync[1], 1u);
            } else {
                while (atomicAdd(&sync[1], 0u) == g)
                    __builtin_amdgcn_s_sleep(1);
            }
        }
        __syncthreads();

        // refresh full h in LDS from the buffer just written
        for (int i = tid; i < Hd; i += GRU_THREADS) h_lds[i] = wbuf[i];
        __syncthreads();
    }

    if (blk == 0)
        for (int i = tid; i < Hd; i += GRU_THREADS) hidden_out[i] = h_lds[i];
}

// out[row] = dot(Mtx[row,:], v)
__global__ __launch_bounds__(256)
void rowdot_kernel(const float* __restrict__ Mtx, const float* __restrict__ v,
                   float* __restrict__ out, int cols) {
    __shared__ float red[256];
    const float* r = Mtx + (size_t)blockIdx.x * cols;
    float acc = 0.0f;
    for (int c = threadIdx.x; c < cols; c += 256) acc += r[c] * v[c];
    red[threadIdx.x] = acc;
    __syncthreads();
    for (int s = 128; s > 0; s >>= 1) {
        if (threadIdx.x < s) red[threadIdx.x] += red[threadIdx.x + s];
        __syncthreads();
    }
    if (threadIdx.x == 0) out[blockIdx.x] = red[0];
}

// p = softmax(s). The per-row term of scores cancels in softmax over axis=1,
// so the whole attention matrix collapses to this one vector.
__global__ __launch_bounds__(1024)
void softmax_vec_kernel(const float* __restrict__ s, float* __restrict__ p, int n) {
    __shared__ float red[1024];
    float m = -INFINITY;
    for (int i = threadIdx.x; i < n; i += 1024) m = fmaxf(m, s[i]);
    red[threadIdx.x] = m;
    __syncthreads();
    for (int st = 512; st > 0; st >>= 1) {
        if (threadIdx.x < st) red[threadIdx.x] = fmaxf(red[threadIdx.x], red[threadIdx.x + st]);
        __syncthreads();
    }
    m = red[0];
    __syncthreads();
    float acc = 0.0f;
    for (int i = threadIdx.x; i < n; i += 1024) acc += __expf(s[i] - m);
    red[threadIdx.x] = acc;
    __syncthreads();
    for (int st = 512; st > 0; st >>= 1) {
        if (threadIdx.x < st) red[threadIdx.x] += red[threadIdx.x + st];
        __syncthreads();
    }
    float inv = 1.0f / red[0];
    for (int i = threadIdx.x; i < n; i += 1024) p[i] = __expf(s[i] - m) * inv;
}

// ctx[h] = sum_s p[s] * enc[s][h]
__global__ __launch_bounds__(256)
void context_kernel(const float* __restrict__ p, const float* __restrict__ enc,
                    float* __restrict__ ctx) {
    int hh = blockIdx.x * 256 + threadIdx.x;
    if (hh >= Hd) return;
    float acc = 0.0f;
    for (int s = 0; s < SSs; ++s) acc += p[s] * enc[(size_t)s * Hd + hh];
    ctx[hh] = acc;
}

// acat(bf16, ST x 2H) = [gru_out | broadcast ctx]
__global__ void acat_kernel(const float* __restrict__ gru,
                            const float* __restrict__ ctx,
                            __bf16* __restrict__ acat) {
    long long n = (long long)STt * 2 * Hd;
    long long stride = (long long)gridDim.x * blockDim.x;
    for (long long i = blockIdx.x * (long long)blockDim.x + threadIdx.x; i < n; i += stride) {
        int t = (int)(i >> 11);
        int c = (int)(i & (2 * Hd - 1));
        float v = (c < Hd) ? gru[(size_t)t * Hd + c] : ctx[c - Hd];
        acat[i] = f2bf(v);
    }
}

// in-place log_softmax per row, ONLINE max+sum (single read pass + write pass)
__global__ __launch_bounds__(256)
void log_softmax_rows_kernel(float* __restrict__ out, int cols) {
    __shared__ float mred[256], sred[256];
    float* row = out + (size_t)blockIdx.x * cols;
    float m = -INFINITY, s = 0.0f;
    for (int c = threadIdx.x; c < cols; c += 256) {
        float xv = row[c];
        if (xv > m) { s = s * __expf(m - xv) + 1.0f; m = xv; }
        else        { s += __expf(xv - m); }
    }
    mred[threadIdx.x] = m; sred[threadIdx.x] = s;
    __syncthreads();
    for (int st = 128; st > 0; st >>= 1) {
        if (threadIdx.x < st) {
            float m1 = mred[threadIdx.x],      s1 = sred[threadIdx.x];
            float m2 = mred[threadIdx.x + st], s2 = sred[threadIdx.x + st];
            float M = fmaxf(m1, m2);
            mred[threadIdx.x] = M;
            sred[threadIdx.x] = s1 * __expf(m1 - M) + s2 * __expf(m2 - M);
        }
        __syncthreads();
    }
    float lse = mred[0] + __logf(sred[0]);
    for (int c = threadIdx.x; c < cols; c += 256) row[c] -= lse;
}

// ---------------------------------------------------------------------------
extern "C" void kernel_launch(void* const* d_in, const int* in_sizes, int n_in,
                              void* d_out, int out_size, void* d_ws, size_t ws_size,
                              hipStream_t stream) {
    const int*   x       = (const int*)  d_in[0];
    const float* enc     = (const float*)d_in[1];
    const float* emb_tab = (const float*)d_in[2];
    const float* w_ih    = (const float*)d_in[3];
    const float* w_hh    = (const float*)d_in[4];
    const float* b_ih    = (const float*)d_in[5];
    const float* b_hh    = (const float*)d_in[6];
    const float* attn_w  = (const float*)d_in[7];
    const float* out_w   = (const float*)d_in[8];
    const float* out_b   = (const float*)d_in[9];
    float* out = (float*)d_out;                 // [ST*V logits | H hidden]

    char* wp = (char*)d_ws;
    auto carve = [&](size_t bytes) -> void* {
        void* p = wp;
        wp += (bytes + 255) & ~(size_t)255;
        return p;
    };
    __bf16*  embB  = (__bf16*)carve((size_t)STt * Ed * 2);        //  2 MB
    __bf16*  wihB  = (__bf16*)carve((size_t)3 * Hd * Ed * 2);     //  3 MB
    float*   gi    = (float*) carve((size_t)STt * 3 * Hd * 4);    // 25 MB
    float*   gruO  = (float*) carve((size_t)STt * Hd * 4);        //  8 MB
    __bf16*  acat  = (__bf16*)carve((size_t)STt * 2 * Hd * 2);    //  8 MB
    __bf16*  outwB = (__bf16*)carve((size_t)Vv * 2 * Hd * 2);     //131 MB
    float*   sB    = (float*) carve((size_t)SSs * 4);
    float*   pvec  = (float*) carve((size_t)SSs * 4);
    float*   ctx0  = (float*) carve((size_t)Hd * 4);
    float*   hbufA = (float*) carve((size_t)Hd * 4);
    float*   hbufB = (float*) carve((size_t)Hd * 4);
    unsigned* sync = (unsigned*)carve(2 * sizeof(unsigned));

    // 1. weight conversions to bf16
    cvt_f32_bf16_kernel<<<2048, 256, 0, stream>>>(w_ih, wihB, (long long)3 * Hd * Ed);
    cvt_f32_bf16_kernel<<<8192, 256, 0, stream>>>(out_w, outwB, (long long)Vv * 2 * Hd);

    // 2. emb = relu(emb_table[x]) -> bf16
    embed_relu_kernel<<<2048, 256, 0, stream>>>(x, emb_tab, embB);

    // 3. gi = emb @ w_ih.T + b_ih   (M=2048, N=3072, K=512)  -- tiled WMMA
    {
        dim3 grid((3 * Hd) / 256, STt / 64);   // (12, 32)
        wmma_gemm_bf16_kernel<<<grid, 128, 0, stream>>>(
            embB, wihB, b_ih, gi, STt, 3 * Hd, Ed, Ed, Ed, 3 * Hd);
    }

    // 4. persistent cooperative GRU scan (w_hh LDS-resident across 32 WGPs)
    gru_sync_init_kernel<<<1, 32, 0, stream>>>(sync);
    {
        size_t smem = (size_t)GRU_ROWS * Hd * 2 + (size_t)Hd * 4 + (size_t)GRU_ROWS * 4;
        gru_persistent_kernel<<<GRU_BLOCKS, GRU_THREADS, smem, stream>>>(
            gi, w_hh, b_hh, gruO, hbufA, hbufB, sync, out + (size_t)STt * Vv);
    }

    // 5. attention collapses to softmax(enc @ w2) -> single context vector
    rowdot_kernel<<<SSs, 256, 0, stream>>>(enc, attn_w + Hd, sB, Hd);
    softmax_vec_kernel<<<1, 1024, 0, stream>>>(sB, pvec, SSs);
    context_kernel<<<Hd / 256, 256, 0, stream>>>(pvec, enc, ctx0);

    // 6. acat = [gru_out | ctx] -> bf16
    acat_kernel<<<4096, 256, 0, stream>>>(gruO, ctx0, acat);

    // 7. logits = acat @ out_w.T + out_b  (M=2048, N=32000, K=2048) -- tiled WMMA
    {
        dim3 grid(Vv / 256, STt / 64);         // (125, 32)
        wmma_gemm_bf16_kernel<<<grid, 128, 0, stream>>>(
            acat, outwB, out_b, out, STt, Vv, 2 * Hd, 2 * Hd, 2 * Hd, Vv);
    }

    // 8. in-place log_softmax per row (online single-pass reduction)
    log_softmax_rows_kernel<<<STt, 256, 0, stream>>>(out, Vv);
}
